// MultiHeadAttention_27891517620602
// MI455X (gfx1250) — compile-verified, matrix operand order checked by reading, not run on hardware
//
#include <hip/hip_runtime.h>

// MHA forward for N=8, S=1024, E=1024, H=16, DK=64 on gfx1250 (wave32, WMMA).
// Pipeline: qkv proj (wmma f16) -> logits+mask -> softmax -> attn@V -> out proj.
// Workspace: 4 x (N*H*S*DK) f16 buffers = 64 MB of d_ws.
// Per-wave tile: 32x64 (8 f32 accumulators), fragments preloaded so each
// k-step issues 8 independent back-to-back v_wmma_f32_16x16x32_f16.

#define NB  8
#define SSQ 1024
#define EEM 1024
#define HHD 16
#define DKK 64

typedef __attribute__((ext_vector_type(16))) _Float16 v16h;
typedef __attribute__((ext_vector_type(8)))  _Float16 h8;
typedef __attribute__((ext_vector_type(8)))  float    v8f;
typedef __attribute__((ext_vector_type(4)))  float    f4;

__device__ __forceinline__ v8f wmma_f16(v16h a, v16h b, v8f c) {
  return __builtin_amdgcn_wmma_f32_16x16x32_f16(
      /*neg_a=*/false, a, /*neg_b=*/false, b,
      /*c_mod=*/(short)0, c, /*reuse_a=*/false, /*reuse_b=*/false);
}

// 16x32 f16 A-fragment (ISA 7.12.2): lane&15 = row, lane bit4 selects the
// K-halves {0..7,16..23} vs {8..15,24..31}. Two contiguous 16B loads.
__device__ __forceinline__ v16h frag_f16(const _Float16* base, int row, int ld,
                                         int k0, int lane) {
  const _Float16* p = base + (size_t)(row + (lane & 15)) * ld + k0 + ((lane >> 4) & 1) * 8;
  union { v16h v; h8 h[2]; } u;
  u.h[0] = *(const h8*)(p);
  u.h[1] = *(const h8*)(p + 16);
  return u.v;
}

// Same fragment sourced from f32 memory with inline f32->f16 convert.
__device__ __forceinline__ v16h frag_f32(const float* base, int row, int ld,
                                         int k0, int lane) {
  const float* p = base + (size_t)(row + (lane & 15)) * ld + k0 + ((lane >> 4) & 1) * 8;
  f4 a = *(const f4*)(p);
  f4 b = *(const f4*)(p + 4);
  f4 c = *(const f4*)(p + 16);
  f4 d = *(const f4*)(p + 20);
  v16h r;
#pragma unroll
  for (int i = 0; i < 4; ++i) {
    r[i]      = (_Float16)a[i];
    r[4 + i]  = (_Float16)b[i];
    r[8 + i]  = (_Float16)c[i];
    r[12 + i] = (_Float16)d[i];
  }
  return r;
}

__device__ __forceinline__ v16h load_a(const float* A, int row, int ld, int k0, int lane) {
  return frag_f32(A, row, ld, k0, lane);
}
__device__ __forceinline__ v16h load_a(const _Float16* A, int row, int ld, int k0, int lane) {
  return frag_f16(A, row, ld, k0, lane);
}

// ---------------------------------------------------------------------------
// Projection GEMM: Y[8192,1024] = A[8192,1024] @ W[1024,1024] + bias.
// MODE 0: f16 out, head layout  [n][h][s][d]   (Q, K)
// MODE 1: f16 out, transposed   [n][h][d][s]   (V, so attn@V B-frags are rows)
// MODE 2: f32 out, row-major (final output into d_out)
// Block = 128 threads (4 waves), 128x64 C tile, wave = 32x64 strip (8 acc).
// W tile (32x64) transposed f32->f16 into LDS with packed ds_store_b128.
// ---------------------------------------------------------------------------
template <typename AT, int MODE>
__global__ __launch_bounds__(128) void proj_kernel(const AT* __restrict__ A,
                                                   const float* __restrict__ W,
                                                   const float* __restrict__ bias,
                                                   void* __restrict__ Out) {
  __shared__ _Float16 Bt[64 * 40];  // [col][k], stride 40 halves (80B, 16B-aligned)
  const int tid  = threadIdx.x;
  const int lane = tid & 31;
  const int wv   = tid >> 5;
  const int m0   = blockIdx.x * 128;
  const int n0   = blockIdx.y * 64;
  const int rowA = m0 + wv * 32;

  v8f acc[8] = {};

  for (int k0 = 0; k0 < EEM; k0 += 32) {
    // Stage the 32x64 W tile transposed: each thread packs 8 column elements
    // into one h8 and issues a single 16B LDS store (2 slots per thread).
#pragma unroll
    for (int it = 0; it < 2; ++it) {
      const int slot = it * 128 + tid;   // 0..255
      const int cc   = slot & 63;        // output column within tile
      const int k8   = (slot >> 6) * 8;  // 0,8,16,24
      h8 pk;
#pragma unroll
      for (int j = 0; j < 8; ++j)
        pk[j] = (_Float16)W[(size_t)(k0 + k8 + j) * EEM + n0 + cc];
      *(h8*)&Bt[cc * 40 + k8] = pk;
    }
    __syncthreads();

    if (k0 + 32 < EEM)
      __builtin_prefetch(A + (size_t)(rowA + (lane & 15)) * EEM + k0 + 32, 0, 1);

    // Preload all fragments, then an uninterrupted 8-deep WMMA chain.
    v16h a0 = load_a(A, rowA, EEM, k0, lane);
    v16h a1 = load_a(A, rowA + 16, EEM, k0, lane);
    v16h bf[4];
#pragma unroll
    for (int ct = 0; ct < 4; ++ct) {
      const _Float16* bp = Bt + (ct * 16 + (lane & 15)) * 40 + ((lane >> 4) & 1) * 8;
      union { v16h v; h8 h[2]; } u;
      u.h[0] = *(const h8*)(bp);
      u.h[1] = *(const h8*)(bp + 16);
      bf[ct] = u.v;
    }
#pragma unroll
    for (int ct = 0; ct < 4; ++ct) acc[ct]     = wmma_f16(a0, bf[ct], acc[ct]);
#pragma unroll
    for (int ct = 0; ct < 4; ++ct) acc[4 + ct] = wmma_f16(a1, bf[ct], acc[4 + ct]);
    __syncthreads();
  }

#pragma unroll
  for (int half = 0; half < 2; ++half) {
#pragma unroll
    for (int ct = 0; ct < 4; ++ct) {
      const int ncol = n0 + ct * 16 + (lane & 15);
      const float bv = bias[ncol];
#pragma unroll
      for (int r = 0; r < 8; ++r) {
        const int m = rowA + half * 16 + ((lane >> 4) & 1) * 8 + r;
        const float val = acc[half * 4 + ct][r] + bv;
        if (MODE == 2) {
          ((float*)Out)[(size_t)m * EEM + ncol] = val;
        } else {
          const int n = m >> 10, s = m & (SSQ - 1);
          const int h = ncol >> 6, d = ncol & (DKK - 1);
          _Float16* o = (_Float16*)Out;
          if (MODE == 0)
            o[(((size_t)(n * HHD + h)) * SSQ + s) * DKK + d] = (_Float16)val;
          else
            o[(((size_t)(n * HHD + h)) * DKK + d) * SSQ + s] = (_Float16)val;
        }
      }
    }
  }
}

// ---------------------------------------------------------------------------
// Logits: per (n,h), L = Qh @ Kh^T / 8, masked -> -1e9, f32 into attn buffer.
// Kh rows ARE Kh^T columns, so B-fragments load straight from kh.
// Block = 128 threads, 128x64 logits tile, wave = 32x64 strip.
// ---------------------------------------------------------------------------
__global__ __launch_bounds__(128) void logits_kernel(const _Float16* __restrict__ qh,
                                                     const _Float16* __restrict__ kh,
                                                     const unsigned char* __restrict__ mask,
                                                     float* __restrict__ attn) {
  const int tid  = threadIdx.x;
  const int lane = tid & 31;
  const int wv   = tid >> 5;
  const int q0   = blockIdx.x * 128 + wv * 32;
  const int t0   = blockIdx.y * 64;
  const int nh   = blockIdx.z;       // n*H + h
  const int n    = nh >> 4;

  const _Float16* qb = qh + (size_t)nh * SSQ * DKK;
  const _Float16* kb = kh + (size_t)nh * SSQ * DKK;

  v8f acc[8] = {};
#pragma unroll
  for (int k0 = 0; k0 < DKK; k0 += 32) {
    v16h a0 = frag_f16(qb, q0, DKK, k0, lane);
    v16h a1 = frag_f16(qb, q0 + 16, DKK, k0, lane);
    v16h bf[4];
#pragma unroll
    for (int ct = 0; ct < 4; ++ct)
      bf[ct] = frag_f16(kb, t0 + ct * 16, DKK, k0, lane);
#pragma unroll
    for (int ct = 0; ct < 4; ++ct) acc[ct]     = wmma_f16(a0, bf[ct], acc[ct]);
#pragma unroll
    for (int ct = 0; ct < 4; ++ct) acc[4 + ct] = wmma_f16(a1, bf[ct], acc[4 + ct]);
  }

  const float scale = 0.125f;  // 1/sqrt(DK)
#pragma unroll
  for (int half = 0; half < 2; ++half) {
#pragma unroll
    for (int ct = 0; ct < 4; ++ct) {
      const int t = t0 + ct * 16 + (lane & 15);
#pragma unroll
      for (int r = 0; r < 8; ++r) {
        const int s = q0 + half * 16 + ((lane >> 4) & 1) * 8 + r;
        const float val = acc[half * 4 + ct][r] * scale;
        const unsigned char mb = mask[((size_t)n * SSQ + s) * SSQ + t];
        attn[(((size_t)nh) * SSQ + s) * SSQ + t] = mb ? -1e9f : val;
      }
    }
  }
}

// ---------------------------------------------------------------------------
// Row softmax over S=1024, in place. One 256-thread block per row.
// ---------------------------------------------------------------------------
__global__ __launch_bounds__(256) void softmax_kernel(float* __restrict__ attn) {
  float* row = attn + (size_t)blockIdx.x * SSQ;
  const int t = threadIdx.x;
  __shared__ float red[256];

  f4 x = *(const f4*)(row + t * 4);
  float m = fmaxf(fmaxf(x[0], x[1]), fmaxf(x[2], x[3]));
  red[t] = m;
  __syncthreads();
  for (int s = 128; s > 0; s >>= 1) {
    if (t < s) red[t] = fmaxf(red[t], red[t + s]);
    __syncthreads();
  }
  const float rmax = red[0];
  __syncthreads();

  f4 e;
  e[0] = __expf(x[0] - rmax);
  e[1] = __expf(x[1] - rmax);
  e[2] = __expf(x[2] - rmax);
  e[3] = __expf(x[3] - rmax);
  red[t] = e[0] + e[1] + e[2] + e[3];
  __syncthreads();
  for (int s = 128; s > 0; s >>= 1) {
    if (t < s) red[t] += red[t + s];
    __syncthreads();
  }
  const float inv = 1.0f / red[0];

  f4 y;
  y[0] = e[0] * inv; y[1] = e[1] * inv; y[2] = e[2] * inv; y[3] = e[3] * inv;
  *(f4*)(row + t * 4) = y;
}

// ---------------------------------------------------------------------------
// attn @ V: per (n,h), Oh[1024,64] = attn[1024,1024](f32->f16) @ Vh.
// B-fragments are contiguous rows of vhT ([n][h][d][s]).
// Block = 128 threads, 128x64 tile, wave = 32x64 strip.
// Output stored f16 head-interleaved as rows of (N*S, E) for the final GEMM.
// ---------------------------------------------------------------------------
__global__ __launch_bounds__(128) void av_kernel(const float* __restrict__ attn,
                                                 const _Float16* __restrict__ vhT,
                                                 _Float16* __restrict__ oflat) {
  const int tid  = threadIdx.x;
  const int lane = tid & 31;
  const int wv   = tid >> 5;
  const int a0r  = blockIdx.x * 128 + wv * 32;
  const int nh   = blockIdx.z;
  const int n    = nh >> 4, h = nh & 15;

  const float*    ab = attn + (size_t)nh * SSQ * SSQ;
  const _Float16* vb = vhT + (size_t)nh * DKK * SSQ;

  v8f acc[8] = {};
  for (int k0 = 0; k0 < SSQ; k0 += 32) {
    v16h a0 = frag_f32(ab, a0r, SSQ, k0, lane);
    v16h a1 = frag_f32(ab, a0r + 16, SSQ, k0, lane);
    v16h bf[4];
#pragma unroll
    for (int ct = 0; ct < 4; ++ct)
      bf[ct] = frag_f16(vb, ct * 16, SSQ, k0, lane);
#pragma unroll
    for (int ct = 0; ct < 4; ++ct) acc[ct]     = wmma_f16(a0, bf[ct], acc[ct]);
#pragma unroll
    for (int ct = 0; ct < 4; ++ct) acc[4 + ct] = wmma_f16(a1, bf[ct], acc[4 + ct]);
  }

#pragma unroll
  for (int half = 0; half < 2; ++half) {
#pragma unroll
    for (int ct = 0; ct < 4; ++ct) {
      const int d = ct * 16 + (lane & 15);
#pragma unroll
      for (int r = 0; r < 8; ++r) {
        const int s = a0r + half * 16 + ((lane >> 4) & 1) * 8 + r;
        oflat[((size_t)n * SSQ + s) * EEM + h * DKK + d] = (_Float16)acc[half * 4 + ct][r];
      }
    }
  }
}

extern "C" void kernel_launch(void* const* d_in, const int* in_sizes, int n_in,
                              void* d_out, int out_size, void* d_ws, size_t ws_size,
                              hipStream_t stream) {
  (void)in_sizes; (void)n_in; (void)out_size; (void)ws_size;

  const float* q  = (const float*)d_in[0];
  const float* k  = (const float*)d_in[1];
  const float* v  = (const float*)d_in[2];
  const unsigned char* mask = (const unsigned char*)d_in[3];  // jnp bool = 1 byte
  const float* wq = (const float*)d_in[4];
  const float* bq = (const float*)d_in[5];
  const float* wk = (const float*)d_in[6];
  const float* bk = (const float*)d_in[7];
  const float* wv = (const float*)d_in[8];
  const float* bv = (const float*)d_in[9];
  const float* wo = (const float*)d_in[10];
  const float* bo = (const float*)d_in[11];

  float* o_out = (float*)d_out;                       // (N,S,E)
  float* attn  = o_out + (size_t)NB * SSQ * EEM;      // (N,H,S,S)

  const size_t HSZ = (size_t)NB * HHD * SSQ * DKK;    // 8M f16 = 16 MB each
  _Float16* qh    = (_Float16*)d_ws;
  _Float16* kh    = qh + HSZ;
  _Float16* vhT   = kh + HSZ;
  _Float16* oflat = vhT + HSZ;                        // total 64 MB of d_ws

  const dim3 pg(NB * SSQ / 128, EEM / 64);            // (64, 16)
  proj_kernel<float, 0><<<pg, 128, 0, stream>>>(q, wq, bq, qh);
  proj_kernel<float, 0><<<pg, 128, 0, stream>>>(k, wk, bk, kh);
  proj_kernel<float, 1><<<pg, 128, 0, stream>>>(v, wv, bv, vhT);

  logits_kernel<<<dim3(SSQ / 128, SSQ / 64, NB * HHD), 128, 0, stream>>>(qh, kh, mask, attn);
  softmax_kernel<<<NB * HHD * SSQ, 256, 0, stream>>>(attn);
  av_kernel<<<dim3(SSQ / 128, 1, NB * HHD), 128, 0, stream>>>(attn, vhT, oflat);

  proj_kernel<_Float16, 2><<<pg, 128, 0, stream>>>(oflat, wo, bo, o_out);
}
